// YOLOScript_6579889897824
// MI455X (gfx1250) — matrix operand order repaired and checked by reading, not compiled, not run on hardware
//
#include <hip/hip_runtime.h>

typedef __attribute__((ext_vector_type(16))) _Float16 v16h;
typedef __attribute__((ext_vector_type(8)))  _Float16 v8h;
typedef __attribute__((ext_vector_type(8)))  float    v8f;
typedef __attribute__((ext_vector_type(4)))  int      v4i;

#define NDET 2535
#define NKEEP 300

// ---------------------------------------------------------------------------
// Pack NCHW f32 input -> NHWC f16, channels zero-padded to Cpad
// ---------------------------------------------------------------------------
__global__ __launch_bounds__(256) void pack_input(const float* __restrict__ img,
                                                  _Float16* __restrict__ dst,
                                                  int N, int H, int W, int C, int Cpad) {
  long long total = (long long)N * H * W;
  long long p = (long long)blockIdx.x * 256 + threadIdx.x;
  if (p >= total) return;
  long long n  = p / ((long long)H * W);
  long long hw = p % ((long long)H * W);
  _Float16* o = dst + p * Cpad;
  for (int c = 0; c < Cpad; ++c) {
    float v = (c < C) ? img[(n * C + c) * (long long)H * W + hw] : 0.0f;
    o[c] = (_Float16)v;
  }
}

// ---------------------------------------------------------------------------
// Pack OIHW f32 weights into WMMA A-fragment layout (16x32 f16 tiles),
// tile-outermost so each out-channel tile's fragments are CONTIGUOUS:
//   idx = (((tile*taps + tap)*nChunks + chunk)*32 + lane)*16 + h
// lane<16 holds K base 0, lane>=16 holds K base 8; vgpr j<4 -> K=2j, j>=4 -> 16+2(j-4)
// ---------------------------------------------------------------------------
__global__ __launch_bounds__(256) void pack_weights(const float* __restrict__ w,
                                                    _Float16* __restrict__ wp,
                                                    int Cout, int Cin, int taps,
                                                    int nChunks, int nTiles) {
  long long total = (long long)taps * nChunks * nTiles * 512;
  long long idx = (long long)blockIdx.x * 256 + threadIdx.x;
  if (idx >= total) return;
  int h    = (int)(idx & 15);
  int lane = (int)((idx >> 4) & 31);
  long long t = idx >> 9;
  int chunk = (int)(t % nChunks); t /= nChunks;
  int tap   = (int)(t % taps);
  int tile  = (int)(t / taps);
  int row = lane & 15;
  int j = h >> 1;
  int kj = (j < 4) ? (2 * j) : (16 + 2 * (j - 4));
  int k = ((lane >> 4) << 3) + kj + (h & 1);
  int ci = chunk * 32 + k;
  int co = tile * 16 + row;
  float v = 0.0f;
  if (co < Cout && ci < Cin)
    v = w[(long long)(co * Cin + ci) * taps + tap];
  wp[idx] = (_Float16)v;
}

// ---------------------------------------------------------------------------
// Cooperative weight staging: async global->LDS DMA when available.
// Builtin signature (from toolchain diagnostic): (v4i addrspace(1)*,
// v4i addrspace(3)*, imm offset, imm cpol).
// ---------------------------------------------------------------------------
__device__ __forceinline__ void stage_weights_lds(const _Float16* wbase, char* smem,
                                                  int total16) {
#if defined(__has_builtin) && __has_builtin(__builtin_amdgcn_global_load_async_to_lds_b128)
  {
    __attribute__((address_space(1))) v4i* gsrc =
        (__attribute__((address_space(1))) v4i*)wbase;
    __attribute__((address_space(3))) v4i* ldst =
        (__attribute__((address_space(3))) v4i*)smem;
    for (int t = threadIdx.x; t < total16; t += 256) {
      __builtin_amdgcn_global_load_async_to_lds_b128(gsrc + t, ldst + t, 0, 0);
    }
#if __has_builtin(__builtin_amdgcn_s_wait_asynccnt)
    __builtin_amdgcn_s_wait_asynccnt(0);
#else
    asm volatile("s_wait_asynccnt 0" ::: "memory");
#endif
  }
#else
  {
    const uint4* gs = (const uint4*)wbase;
    uint4* ld = (uint4*)smem;
    for (int t = threadIdx.x; t < total16; t += 256) ld[t] = gs[t];
  }
#endif
  __syncthreads();
}

// ---------------------------------------------------------------------------
// Implicit-GEMM conv (3x3 stride-2 SAME, pad_before=0) via
// v_wmma_f32_16x16x32_f16. NHWC f16 in / NHWC f16 out, fused bias + leaky.
// Wave: 16 out-channels x 32 pixels (two N tiles reuse one A-frag).
// Block: 8 waves; per-tile weights staged in LDS once per block.
// ---------------------------------------------------------------------------
__global__ __launch_bounds__(256) void conv_wmma(const _Float16* __restrict__ in,
                                                 const _Float16* __restrict__ wp,
                                                 const float* __restrict__ bias,
                                                 _Float16* __restrict__ out,
                                                 int inH, int inW, int Cpad,
                                                 int outH, int outW, int Cout,
                                                 int stride, int is3x3,
                                                 int nChunks, int nTiles, int relu) {
  extern __shared__ char smem[];
  int lane = threadIdx.x & 31;
  int wave = threadIdx.x >> 5;
  int n = blockIdx.z;
  int ocTile = blockIdx.y;
  int taps = is3x3 ? 9 : 1;
  int fragCnt = taps * nChunks;

  // stage this tile's packed weights (contiguous) into LDS
  const _Float16* wbase = wp + (long long)ocTile * fragCnt * 512;
  stage_weights_lds(wbase, smem, fragCnt * 64);

  int outHW = outH * outW;
  int pixBase = (blockIdx.x * 8 + wave) * 32;
  int p0 = pixBase + (lane & 15);
  int p1 = p0 + 16;
  bool pv0 = p0 < outHW, pv1 = p1 < outHW;
  int pc0 = pv0 ? p0 : 0, pc1 = pv1 ? p1 : 0;
  int oy0 = pc0 / outW, ox0 = pc0 % outW;
  int oy1 = pc1 / outW, ox1 = pc1 % outW;
  int chanBase = (lane >> 4) << 3;  // 0 or 8 (B-frag K base; second half at +16)

  v8f acc0 = {}, acc1 = {};
  for (int tap = 0; tap < taps; ++tap) {
    int ky = is3x3 ? tap / 3 : 0;
    int kx = is3x3 ? tap % 3 : 0;
    int iy0 = oy0 * stride + ky, ix0 = ox0 * stride + kx;
    int iy1 = oy1 * stride + ky, ix1 = ox1 * stride + kx;
    bool v0 = pv0 && (iy0 < inH) && (ix0 < inW);
    bool v1 = pv1 && (iy1 < inH) && (ix1 < inW);
    const _Float16* s0 = in + ((((long long)n * inH + iy0) * inW + ix0) * Cpad) + chanBase;
    const _Float16* s1 = in + ((((long long)n * inH + iy1) * inW + ix1) * Cpad) + chanBase;
    for (int ch = 0; ch < nChunks; ++ch) {
      v16h a = *(const v16h*)(smem + ((size_t)((tap * nChunks + ch) * 32) + lane) * 32);
      v16h b0 = {};
      if (v0) b0 = __builtin_shufflevector(*(const v8h*)(s0 + ch * 32), *(const v8h*)(s0 + ch * 32 + 16), 0, 1, 2, 3, 4, 5, 6, 7, 8, 9, 10, 11, 12, 13, 14, 15);
      v16h b1 = {};
      if (v1) b1 = __builtin_shufflevector(*(const v8h*)(s1 + ch * 32), *(const v8h*)(s1 + ch * 32 + 16), 0, 1, 2, 3, 4, 5, 6, 7, 8, 9, 10, 11, 12, 13, 14, 15);
      if (ch + 1 < nChunks) __builtin_prefetch(s0 + (ch + 1) * 32);  // global_prefetch_b8
      acc0 = __builtin_amdgcn_wmma_f32_16x16x32_f16(false, a, false, b0,
                                                    (short)0, acc0, false, false);
      acc1 = __builtin_amdgcn_wmma_f32_16x16x32_f16(false, a, false, b1,
                                                    (short)0, acc1, false, false);
    }
  }

  int ocOff = ocTile * 16 + ((lane >> 4) << 3);  // D: lane<16 -> rows 0..7, else 8..15
  if (pv0) {
    v8h ov;
#pragma unroll
    for (int r = 0; r < 8; ++r) {
      float v = acc0[r] + bias[ocOff + r];
      if (relu) v = v > 0.0f ? v : 0.1f * v;
      ov[r] = (_Float16)v;
    }
    *(v8h*)(out + (((long long)n * outH + oy0) * outW + ox0) * Cout + ocOff) = ov;
  }
  if (pv1) {
    v8h ov;
#pragma unroll
    for (int r = 0; r < 8; ++r) {
      float v = acc1[r] + bias[ocOff + r];
      if (relu) v = v > 0.0f ? v : 0.1f * v;
      ov[r] = (_Float16)v;
    }
    *(v8h*)(out + (((long long)n * outH + oy1) * outW + ox1) * Cout + ocOff) = ov;
  }
}

// ---------------------------------------------------------------------------
// Head: 1x1 conv via WMMA + fused YOLO decode, writes pred[n][row][85] f32.
// Cout padded to 256 (16 tiles); real channels c<255, c = a*85 + e.
// ---------------------------------------------------------------------------
struct Anch { float w0, h0, w1, h1, w2, h2; };

__device__ inline float sigm(float v) { return 1.0f / (1.0f + expf(-v)); }

__global__ __launch_bounds__(256) void head_wmma(const _Float16* __restrict__ in,
                                                 const _Float16* __restrict__ wp,
                                                 const float* __restrict__ bias,
                                                 float* __restrict__ pred,
                                                 int side, int Cpad, int nChunks,
                                                 int rowOffset, Anch anch) {
  int lane = threadIdx.x & 31;
  int wave = threadIdx.x >> 5;
  int n = blockIdx.z;
  int ocTile = blockIdx.y;
  int outHW = side * side;
  int pixBase = (blockIdx.x * 8 + wave) * 16;
  int p = pixBase + (lane & 15);
  bool pv = p < outHW;
  int pc = pv ? p : 0;
  int oy = pc / side, ox = pc % side;
  int chanBase = (lane >> 4) << 3;

  v8f acc = {};
  const _Float16* src = in + (((long long)n * outHW + pc) * Cpad) + chanBase;
  for (int ch = 0; ch < nChunks; ++ch) {
    v16h b = {};
    if (pv) b = __builtin_shufflevector(*(const v8h*)(src + ch * 32), *(const v8h*)(src + ch * 32 + 16), 0, 1, 2, 3, 4, 5, 6, 7, 8, 9, 10, 11, 12, 13, 14, 15);
    // tile-outermost layout, taps==1
    const _Float16* wl = wp + (((long long)ocTile * nChunks + ch) * 32 + lane) * 16;
    v16h a = *(const v16h*)wl;
    acc = __builtin_amdgcn_wmma_f32_16x16x32_f16(false, a, false, b,
                                                 (short)0, acc, false, false);
  }

  if (pv) {
    int ocOff = ocTile * 16 + ((lane >> 4) << 3);
#pragma unroll
    for (int r = 0; r < 8; ++r) {
      int c = ocOff + r;
      if (c < 255) {
        float v = acc[r] + bias[c];
        int a = c / 85, e = c - a * 85;
        float aw = (a == 0) ? anch.w0 : (a == 1) ? anch.w1 : anch.w2;
        float ah = (a == 0) ? anch.h0 : (a == 1) ? anch.h1 : anch.h2;
        float res;
        if (e == 0)      res = (sigm(v) + (float)ox) / (float)side;
        else if (e == 1) res = (sigm(v) + (float)oy) / (float)side;
        else if (e == 2) res = expf(v) * aw / 416.0f;
        else if (e == 3) res = expf(v) * ah / 416.0f;
        else             res = sigm(v);
        long long row = rowOffset + (long long)a * outHW + pc;
        pred[((long long)n * NDET + row) * 85 + e] = res;
      }
    }
  }
}

// ---------------------------------------------------------------------------
// Per-image top-300 + NMS. One block per image.
// ---------------------------------------------------------------------------
__global__ __launch_bounds__(256) void nms_kernel(const float* __restrict__ pred,
                                                  float* __restrict__ out) {
  int n = blockIdx.x;
  int tid = threadIdx.x;
  __shared__ float s_score[NDET];
  __shared__ float s_cc[NDET];
  __shared__ short s_lab[NDET];
  __shared__ float rbV[256];
  __shared__ int   rbI[256];
  __shared__ float selBox[NKEEP][4];
  __shared__ float selObj[NKEEP];
  __shared__ float selCC[NKEEP];
  __shared__ int   selLab[NKEEP];
  __shared__ int   keep[NKEEP];
  __shared__ int   s_any;

  const float* P = pred + (long long)n * NDET * 85;
  for (int i = tid; i < NDET; i += 256) {
    const float* pr = P + (long long)i * 85;
    float obj = pr[4];
    float mc = pr[5];
    int lb = 0;
    for (int c = 1; c < 80; ++c) {
      float v = pr[5 + c];
      if (v > mc) { mc = v; lb = c; }
    }
    float s = obj * mc;
    if (!(s >= 0.1f)) s = 0.0f;
    s_score[i] = s;
    s_cc[i] = mc;
    s_lab[i] = (short)lb;
  }
  __syncthreads();

  // iterative block-wide argmax: top-300 (stable tie-break on smaller index)
  for (int k = 0; k < NKEEP; ++k) {
    float bv = -1.0f;
    int bi = NDET;
    for (int i = tid; i < NDET; i += 256) {
      float v = s_score[i];
      if (v > bv || (v == bv && i < bi)) { bv = v; bi = i; }
    }
    rbV[tid] = bv;
    rbI[tid] = bi;
    __syncthreads();
    for (int off = 128; off > 0; off >>= 1) {
      if (tid < off) {
        float v2 = rbV[tid + off];
        int   i2 = rbI[tid + off];
        if (v2 > rbV[tid] || (v2 == rbV[tid] && i2 < rbI[tid])) {
          rbV[tid] = v2; rbI[tid] = i2;
        }
      }
      __syncthreads();
    }
    if (tid == 0) {
      int idx = rbI[0];
      float sc = rbV[0];
      if (idx >= NDET) idx = 0;
      const float* pr = P + (long long)idx * 85;
      float bx = pr[0], by = pr[1], bw = pr[2], bh = pr[3];
      selBox[k][0] = (by - bh * 0.5f) * 416.0f;  // y1
      selBox[k][1] = (bx - bw * 0.5f) * 416.0f;  // x1
      selBox[k][2] = (by + bh * 0.5f) * 416.0f;  // y2
      selBox[k][3] = (bx + bw * 0.5f) * 416.0f;  // x2
      selObj[k] = pr[4];
      selCC[k]  = s_cc[idx];
      selLab[k] = (int)s_lab[idx];
      keep[k]   = (sc > 0.0f) ? 1 : 0;
      s_score[idx] = -1.0f;
    }
    __syncthreads();
  }

  // sequential suppression (matches reference fori_loop semantics)
  for (int i = 1; i < NKEEP; ++i) {
    if (tid == 0) s_any = 0;
    __syncthreads();
    if (keep[i]) {
      float y1 = selBox[i][0], x1 = selBox[i][1];
      float y2 = selBox[i][2], x2 = selBox[i][3];
      float ai = (y2 - y1) * (x2 - x1);
      int f = 0;
      for (int j = tid; j < i; j += 256) {
        if (keep[j] && selLab[j] == selLab[i]) {
          float ty = fmaxf(selBox[j][0], y1);
          float tx = fmaxf(selBox[j][1], x1);
          float by2 = fminf(selBox[j][2], y2);
          float bx2 = fminf(selBox[j][3], x2);
          float hh = fmaxf(by2 - ty, 0.0f);
          float ww = fmaxf(bx2 - tx, 0.0f);
          float inter = hh * ww;
          float aj = (selBox[j][2] - selBox[j][0]) * (selBox[j][3] - selBox[j][1]);
          float iou = inter / (ai + aj - inter + 1e-9f);
          if (iou > 0.5f) f = 1;
        }
      }
      if (f) atomicOr(&s_any, 1);
    }
    __syncthreads();
    if (tid == 0 && s_any) keep[i] = 0;
    __syncthreads();
  }

  for (int k = tid; k < NKEEP; k += 256) {
    float* o = out + ((long long)n * NKEEP + k) * 7;
    if (keep[k]) {
      o[0] = selBox[k][0]; o[1] = selBox[k][1];
      o[2] = selBox[k][2]; o[3] = selBox[k][3];
      o[4] = selObj[k];    o[5] = selCC[k];
      o[6] = (float)selLab[k];
    } else {
      for (int t = 0; t < 7; ++t) o[t] = 0.0f;
    }
  }
}

// ---------------------------------------------------------------------------
// Host side
// ---------------------------------------------------------------------------
static inline size_t alignup(size_t x) { return (x + 255) & ~(size_t)255; }

extern "C" void kernel_launch(void* const* d_in, const int* in_sizes, int n_in,
                              void* d_out, int out_size, void* d_ws, size_t ws_size,
                              hipStream_t stream) {
  const float* images = (const float*)d_in[0];
  const float* w1 = (const float*)d_in[1];  const float* b1 = (const float*)d_in[2];
  const float* w2 = (const float*)d_in[3];  const float* b2 = (const float*)d_in[4];
  const float* w3 = (const float*)d_in[5];  const float* b3 = (const float*)d_in[6];
  const float* w4 = (const float*)d_in[7];  const float* b4 = (const float*)d_in[8];
  const float* w5 = (const float*)d_in[9];  const float* b5 = (const float*)d_in[10];
  const float* wh1 = (const float*)d_in[11]; const float* bh1 = (const float*)d_in[12];
  const float* wh2 = (const float*)d_in[13]; const float* bh2 = (const float*)d_in[14];

  char* ws = (char*)d_ws;
  size_t off = 0;
  auto suballoc = [&](size_t bytes) -> char* {
    char* p = ws + off;
    off += alignup(bytes);
    return p;
  };

  const int N = 8;
  _Float16* x0 = (_Float16*)suballoc((size_t)N * 416 * 416 * 32 * 2);
  _Float16* a1 = (_Float16*)suballoc((size_t)N * 208 * 208 * 32 * 2);
  _Float16* a2 = (_Float16*)suballoc((size_t)N * 104 * 104 * 64 * 2);
  _Float16* a3 = (_Float16*)suballoc((size_t)N * 52 * 52 * 128 * 2);
  _Float16* a4 = (_Float16*)suballoc((size_t)N * 26 * 26 * 256 * 2);
  _Float16* a5 = (_Float16*)suballoc((size_t)N * 13 * 13 * 512 * 2);
  float*  pred = (float*)suballoc((size_t)N * NDET * 85 * 4);
  _Float16* wp1 = (_Float16*)suballoc((size_t)9 * 1 * 2 * 512 * 2);
  _Float16* wp2 = (_Float16*)suballoc((size_t)9 * 1 * 4 * 512 * 2);
  _Float16* wp3 = (_Float16*)suballoc((size_t)9 * 2 * 8 * 512 * 2);
  _Float16* wp4 = (_Float16*)suballoc((size_t)9 * 4 * 16 * 512 * 2);
  _Float16* wp5 = (_Float16*)suballoc((size_t)9 * 8 * 32 * 512 * 2);
  _Float16* wph2 = (_Float16*)suballoc((size_t)1 * 8 * 16 * 512 * 2);
  _Float16* wph1 = (_Float16*)suballoc((size_t)1 * 16 * 16 * 512 * 2);

  // 1) input NCHW f32 -> NHWC f16 (C padded 3 -> 32)
  {
    long long total = (long long)N * 416 * 416;
    int blocks = (int)((total + 255) / 256);
    pack_input<<<blocks, 256, 0, stream>>>(images, x0, N, 416, 416, 3, 32);
  }

  // 2) pack weights into A-fragment layout (tile-outermost)
  auto packW = [&](const float* w, _Float16* wp, int Cout, int Cin, int taps,
                   int nChunks, int nTiles) {
    long long total = (long long)taps * nChunks * nTiles * 512;
    int blocks = (int)((total + 255) / 256);
    pack_weights<<<blocks, 256, 0, stream>>>(w, wp, Cout, Cin, taps, nChunks, nTiles);
  };
  packW(w1, wp1, 32, 3, 9, 1, 2);
  packW(w2, wp2, 64, 32, 9, 1, 4);
  packW(w3, wp3, 128, 64, 9, 2, 8);
  packW(w4, wp4, 256, 128, 9, 4, 16);
  packW(w5, wp5, 512, 256, 9, 8, 32);
  packW(wh2, wph2, 255, 256, 1, 8, 16);
  packW(wh1, wph1, 255, 512, 1, 16, 16);

  // 3) conv stack (3x3 stride-2 SAME, leaky relu)
  auto conv = [&](const _Float16* in, const _Float16* wp, const float* bias,
                  _Float16* out, int inH, int Cpad, int outH, int Cout,
                  int nChunks, int nTiles) {
    int outHW = outH * outH;
    int pixTiles = (outHW + 255) / 256;  // 32 pixels per wave, 8 waves
    dim3 grid(pixTiles, nTiles, N);
    size_t shbytes = (size_t)9 * nChunks * 1024;  // fragCnt KB in LDS
    conv_wmma<<<grid, 256, shbytes, stream>>>(in, wp, bias, out, inH, inH, Cpad,
                                              outH, outH, Cout, 2, 1, nChunks, nTiles, 1);
  };
  conv(x0, wp1, b1, a1, 416, 32, 208, 32, 1, 2);
  conv(a1, wp2, b2, a2, 208, 32, 104, 64, 1, 4);
  conv(a2, wp3, b3, a3, 104, 64, 52, 128, 2, 8);
  conv(a3, wp4, b4, a4, 52, 128, 26, 256, 4, 16);

  // head2 on a4 (26x26), anchors MASK2 = (10,14),(23,27),(37,58), rows 507..
  {
    Anch a = {10.0f, 14.0f, 23.0f, 27.0f, 37.0f, 58.0f};
    int pixTiles = (676 + 127) / 128;
    dim3 grid(pixTiles, 16, N);
    head_wmma<<<grid, 256, 0, stream>>>(a4, wph2, bh2, pred, 26, 256, 8, 507, a);
  }

  conv(a4, wp5, b5, a5, 26, 256, 13, 512, 8, 32);

  // head1 on a5 (13x13), anchors MASK1 = (81,82),(135,169),(344,319), rows 0..506
  {
    Anch a = {81.0f, 82.0f, 135.0f, 169.0f, 344.0f, 319.0f};
    int pixTiles = (169 + 127) / 128;
    dim3 grid(pixTiles, 16, N);
    head_wmma<<<grid, 256, 0, stream>>>(a5, wph1, bh1, pred, 13, 512, 16, 0, a);
  }

  // 4) per-image top-300 + NMS
  nms_kernel<<<N, 256, 0, stream>>>(pred, (float*)d_out);
}
